// SelfAttention_18940805775435
// MI455X (gfx1250) — compile-verified, matrix operand order checked by reading, not run on hardware
//
#include <hip/hip_runtime.h>

// ---------------------------------------------------------------------------
// Problem constants (from reference): B=2, S=2048, D=2048, H=16, KVH=4, HD=128
// ---------------------------------------------------------------------------
#define B_    2
#define S_    2048
#define D_    2048
#define H_    16
#define KVH_  4
#define HD_   128
#define NREP_ 4
#define MTOT  (B_ * S_)          // 4096 rows for all projections

typedef __attribute__((ext_vector_type(16))) __bf16 v16bf;
typedef __attribute__((ext_vector_type(8)))  float  v8f;
typedef unsigned int u32x4 __attribute__((ext_vector_type(4)));
typedef unsigned int u32x8 __attribute__((ext_vector_type(8)));

// round-to-nearest-even fp32 -> bf16 (stored as u16)
__device__ __forceinline__ unsigned short f32_to_bf16(float f) {
  unsigned int u = __float_as_uint(f);
  u += 0x7FFFu + ((u >> 16) & 1u);
  return (unsigned short)(u >> 16);
}
__device__ __forceinline__ float bf16_to_f32(unsigned short h) {
  return __uint_as_float(((unsigned int)h) << 16);
}

// ---------------------------------------------------------------------------
// CDNA5 async global->LDS copy (ASYNCcnt path, ISA 15.18.3 op 98).
// ---------------------------------------------------------------------------
__device__ __forceinline__ void async_copy16(unsigned short* ldsDst,
                                             const unsigned short* gSrc) {
  unsigned int lds = (unsigned int)(unsigned long long)(void*)ldsDst;
  asm volatile("global_load_async_to_lds_b128 %0, %1, off"
               :: "v"(lds), "v"(gSrc) : "memory");
}
__device__ __forceinline__ void wait_async0() {
  asm volatile("s_wait_asynccnt 0" ::: "memory");
}

// ---------------------------------------------------------------------------
// CDNA5 Tensor Data Mover: 2D tile load global->LDS (TENSORcnt path).
// D# per CDNA5 ISA: group0 = {count=1, lds_addr,
// global_addr, type=2}; group1 = {data_size=2B, tensor dims = tile dims
// (no OOB), tile dims, row stride}. Rows are packed contiguously in LDS.
// Issue from ONE wave; EXEC is ignored by TDM.
// ---------------------------------------------------------------------------
__device__ __forceinline__ void tdm_load_tile_2d(
    const void* gaddr, unsigned int lds_addr,
    unsigned int tileW /*elems*/, unsigned int tileH /*rows*/,
    unsigned int rowStrideElems) {
  const unsigned long long ga = (unsigned long long)gaddr;
  u32x4 g0;
  g0[0] = 1u;                                             // count=1 (user D#)
  g0[1] = lds_addr;                                       // LDS byte address
  g0[2] = (unsigned int)ga;                               // global_addr[31:0]
  g0[3] = (unsigned int)((ga >> 32) & 0x01FFFFFFu) | (2u << 30);  // type=2
  u32x8 g1;
  g1[0] = 1u << 16;                                       // data_size=1 (2B)
  g1[1] = (tileW & 0xFFFFu) << 16;                        // tensor_dim0 lo
  g1[2] = (tileW >> 16) | ((tileH & 0xFFFFu) << 16);      // dim0 hi | dim1 lo
  g1[3] = (tileH >> 16) | ((tileW & 0xFFFFu) << 16);      // dim1 hi | tile_dim0
  g1[4] = tileH & 0xFFFFu;                                // tile_dim1 (dim2=0)
  g1[5] = rowStrideElems;                                 // dim0_stride lo
  g1[6] = 0u;                                             // stride hi / dim1_stride
  g1[7] = 0u;
  asm volatile("tensor_load_to_lds %0, %1" :: "s"(g0), "s"(g1) : "memory");
}

union FragCast { uint4 u[2]; v16bf v; };

// Load one 16x32 bf16 WMMA operand fragment (A row-major M x K, or B as
// N-major rows of K), per CDNA5 ISA 7.12.2:
//   lanes 0-15  : row = lane,     VGPR0-3 = K 0..7,   VGPR4-7 = K 16..23
//   lanes 16-31 : row = lane-16,  VGPR0-3 = K 8..15,  VGPR4-7 = K 24..31
__device__ __forceinline__ v16bf load_frag(const unsigned short* base,
                                           int strideElems, int lane) {
  const int g = (lane >> 4) & 1;
  const int r = lane & 15;
  const unsigned short* row = base + r * strideElems;
  FragCast f;
  f.u[0] = *reinterpret_cast<const uint4*>(row + 8 * g);
  f.u[1] = *reinterpret_cast<const uint4*>(row + 16 + 8 * g);
  return f.v;
}

// ---------------------------------------------------------------------------
// Pre-passes: one-time bf16 conversion / weight transpose (bandwidth-trivial)
// ---------------------------------------------------------------------------
__global__ __launch_bounds__(256) void convert_f32_bf16_vec4(
    const float* __restrict__ src, unsigned short* __restrict__ dst) {
  const int i = blockIdx.x * 256 + threadIdx.x;     // one float4 per thread
  const float4 v = reinterpret_cast<const float4*>(src)[i];
  ushort4 o;
  o.x = f32_to_bf16(v.x); o.y = f32_to_bf16(v.y);
  o.z = f32_to_bf16(v.z); o.w = f32_to_bf16(v.w);
  reinterpret_cast<ushort4*>(dst)[i] = o;
}

// Wt[n][k] = bf16(W[k][n]);  W is (K,N) row-major, K,N multiples of 32.
__global__ __launch_bounds__(256) void transpose_w_bf16(
    const float* __restrict__ W, unsigned short* __restrict__ Wt, int K, int N) {
  __shared__ float t[32][33];
  const int n0 = blockIdx.x * 32, k0 = blockIdx.y * 32;
  const int tx = threadIdx.x, ty = threadIdx.y;     // (32, 8)
#pragma unroll
  for (int j = 0; j < 32; j += 8)
    t[ty + j][tx] = W[(size_t)(k0 + ty + j) * N + n0 + tx];
  __syncthreads();
#pragma unroll
  for (int j = 0; j < 32; j += 8)
    Wt[(size_t)(n0 + ty + j) * K + k0 + tx] = f32_to_bf16(t[tx][ty + j]);
}

// ---------------------------------------------------------------------------
// GEMM: C(MxN) = A(MxK,bf16) * Bt(NxK,bf16)^T, f32 accumulate.
// Block tile 256x128x32, 256 threads = 8 waves (4x2), wave tile 64x64.
// Double-buffered LDS fed by the Tensor Data Mover: wave 0 issues ONE
// tensor_load_to_lds per tile, 16 WMMAs run on the current buffer,
// s_wait_tensorcnt 0 + barrier, swap.
// OUT_BF16 selects bf16 or f32 output.
// ---------------------------------------------------------------------------
template <bool OUT_BF16>
__global__ __launch_bounds__(256) void gemm_bf16wmma(
    const unsigned short* __restrict__ A, const unsigned short* __restrict__ Bt,
    void* __restrict__ Cout, int M, int N, int K) {
  __shared__ __align__(16) unsigned short As[2][256][32];  // 2 x 16KB
  __shared__ __align__(16) unsigned short Bs[2][128][32];  // 2 x  8KB

  const int tid  = threadIdx.x;
  const int lane = tid & 31;
  const int wave = tid >> 5;
  const int wm   = wave >> 1;   // 0..3 -> 64 rows each
  const int wn   = wave & 1;    // 0..1 -> 64 cols each
  const int m0   = blockIdx.y * 256;
  const int n0   = blockIdx.x * 128;

  v8f acc[4][4];
#pragma unroll
  for (int i = 0; i < 4; ++i)
#pragma unroll
    for (int j = 0; j < 4; ++j) acc[i][j] = {};

  // TDM-stage one 32-wide K tile into buffer `buf` (wave 0 only)
  auto stage = [&](int buf, int k0) {
    if (wave == 0) {
      tdm_load_tile_2d(&A[(size_t)m0 * K + k0],
                       (unsigned int)(unsigned long long)(void*)&As[buf][0][0],
                       32, 256, (unsigned int)K);
      tdm_load_tile_2d(&Bt[(size_t)n0 * K + k0],
                       (unsigned int)(unsigned long long)(void*)&Bs[buf][0][0],
                       32, 128, (unsigned int)K);
    }
  };

  stage(0, 0);
  if (wave == 0) __builtin_amdgcn_s_wait_tensorcnt(0);
  __syncthreads();

  int buf = 0;
  for (int k0 = 0; k0 < K; k0 += 32) {
    if (k0 + 32 < K) stage(buf ^ 1, k0 + 32);        // DMA next tile

    v16bf a[4], b[4];
#pragma unroll
    for (int i = 0; i < 4; ++i)
      a[i] = load_frag(&As[buf][wm * 64 + i * 16][0], 32, lane);
#pragma unroll
    for (int j = 0; j < 4; ++j)
      b[j] = load_frag(&Bs[buf][wn * 64 + j * 16][0], 32, lane);
#pragma unroll
    for (int i = 0; i < 4; ++i)
#pragma unroll
      for (int j = 0; j < 4; ++j)
        acc[i][j] = __builtin_amdgcn_wmma_f32_16x16x32_bf16(
            false, a[i], false, b[j], (short)0, acc[i][j], false, false);

    if (wave == 0) __builtin_amdgcn_s_wait_tensorcnt(0);
    __syncthreads();
    buf ^= 1;
  }

  // C/D layout: lane col = lane&15, VGPR v -> row v + 8*(lane>>4)
  const int g = lane >> 4, cc = lane & 15;
#pragma unroll
  for (int i = 0; i < 4; ++i)
#pragma unroll
    for (int j = 0; j < 4; ++j)
#pragma unroll
      for (int v = 0; v < 8; ++v) {
        const size_t row = m0 + wm * 64 + i * 16 + v + 8 * g;
        const size_t col = n0 + wn * 64 + j * 16 + cc;
        if (OUT_BF16)
          ((unsigned short*)Cout)[row * N + col] = f32_to_bf16(acc[i][j][v]);
        else
          ((float*)Cout)[row * N + col] = acc[i][j][v];
      }
}

// ---------------------------------------------------------------------------
// RoPE + layout transforms (bf16 in -> bf16 attention layouts)
// ---------------------------------------------------------------------------
__global__ __launch_bounds__(256) void rope_q_kernel(
    const unsigned short* __restrict__ q, const float* __restrict__ fc,
    const float* __restrict__ fs, unsigned short* __restrict__ Qb) {
  int idx = blockIdx.x * 256 + threadIdx.x;      // B*S*H*(HD/2)
  const int d2 = idx & 63; idx >>= 6;
  const int h  = idx & (H_ - 1); idx >>= 4;
  const int s  = idx & (S_ - 1);
  const int b  = idx >> 11;
  const unsigned short* src = q + (((size_t)b * S_ + s) * H_ + h) * HD_;
  const float xr = bf16_to_f32(src[2 * d2]), xi = bf16_to_f32(src[2 * d2 + 1]);
  const float cs = fc[s * 64 + d2], sn = fs[s * 64 + d2];
  unsigned short* dst = Qb + (((size_t)b * H_ + h) * S_ + s) * HD_;
  dst[2 * d2]     = f32_to_bf16(xr * cs - xi * sn);
  dst[2 * d2 + 1] = f32_to_bf16(xr * sn + xi * cs);
}

__global__ __launch_bounds__(256) void rope_k_kernel(
    const unsigned short* __restrict__ k, const float* __restrict__ fc,
    const float* __restrict__ fs, unsigned short* __restrict__ Kb) {
  int idx = blockIdx.x * 256 + threadIdx.x;      // B*S*KVH*(HD/2)
  const int d2 = idx & 63; idx >>= 6;
  const int kh = idx & (KVH_ - 1); idx >>= 2;
  const int s  = idx & (S_ - 1);
  const int b  = idx >> 11;
  const unsigned short* src = k + (((size_t)b * S_ + s) * KVH_ + kh) * HD_;
  const float xr = bf16_to_f32(src[2 * d2]), xi = bf16_to_f32(src[2 * d2 + 1]);
  const float cs = fc[s * 64 + d2], sn = fs[s * 64 + d2];
  unsigned short* dst = Kb + (((size_t)b * KVH_ + kh) * S_ + s) * HD_;
  dst[2 * d2]     = f32_to_bf16(xr * cs - xi * sn);
  dst[2 * d2 + 1] = f32_to_bf16(xr * sn + xi * cs);
}

__global__ __launch_bounds__(256) void v_transpose_kernel(
    const unsigned short* __restrict__ v, unsigned short* __restrict__ Vt) {
  int idx = blockIdx.x * 256 + threadIdx.x;      // B*S*KVH*HD
  const int d  = idx & (HD_ - 1); idx >>= 7;
  const int kh = idx & (KVH_ - 1); idx >>= 2;
  const int s  = idx & (S_ - 1);
  const int b  = idx >> 11;
  Vt[(((size_t)b * KVH_ + kh) * HD_ + d) * S_ + s] =
      v[(((size_t)b * S_ + s) * KVH_ + kh) * HD_ + d];
}

// ---------------------------------------------------------------------------
// Flash-style GQA attention, no mask, full softmax over S keys.
// Block = (b, h, 128 q rows); 8 waves, each wave owns 16 q rows -> softmax
// reductions stay inside 16-lane groups (wave32 shfl_xor only).
// KV tile = 64 keys, staged via async global->LDS copies (ASYNCcnt).
// LDS: K 16KB + V^T 16KB + per-wave P scratch 16KB = 48KB.
// ---------------------------------------------------------------------------
__global__ __launch_bounds__(256) void attn_kernel(
    const unsigned short* __restrict__ Qb,   // (B,H,S,HD)    bf16
    const unsigned short* __restrict__ Kb,   // (B,KVH,S,HD)  bf16
    const unsigned short* __restrict__ Vt,   // (B,KVH,HD,S)  bf16
    unsigned short* __restrict__ Oout) {     // (B,S,H,HD)    bf16
  __shared__ __align__(16) unsigned short Ks[64][128];     // [t][d]
  __shared__ __align__(16) unsigned short Vs[128][64];     // [d][t]
  __shared__ __align__(16) unsigned short Ps[8][16][64];   // per-wave P tile

  const int tid  = threadIdx.x;
  const int lane = tid & 31;
  const int wave = tid >> 5;
  const int qt   = blockIdx.x;          // 0..S/128-1
  const int h    = blockIdx.y;          // 0..H-1
  const int b    = blockIdx.z;          // 0..B-1
  const int kvh  = h >> 2;              // GQA: 4 q heads per kv head
  const int g    = lane >> 4;
  const float scale = 0.08838834764831845f;   // 1/sqrt(128)

  const unsigned short* Qbase =
      Qb + (((size_t)b * H_ + h) * S_ + qt * 128 + wave * 16) * HD_;
  const unsigned short* Kbase = Kb + (((size_t)b * KVH_ + kvh) * S_) * HD_;
  const unsigned short* Vbase = Vt + (((size_t)b * KVH_ + kvh) * HD_) * S_;

  // Q fragments for the wave's 16 rows: 4 K-slices of HD, held all kernel.
  v16bf qf[4];
#pragma unroll
  for (int kk = 0; kk < 4; ++kk) qf[kk] = load_frag(Qbase + kk * 32, HD_, lane);

  v8f o[8];
#pragma unroll
  for (int j = 0; j < 8; ++j) o[j] = {};
  float mrow[8], lrow[8];
#pragma unroll
  for (int v = 0; v < 8; ++v) { mrow[v] = -3.402823466e38f; lrow[v] = 0.f; }

  for (int t0 = 0; t0 < S_; t0 += 64) {
    // async-stage K tile (64 x 128) and V^T tile (128 x 64)
#pragma unroll
    for (int i = tid; i < 64 * 16; i += 256) {
      const int t = i >> 4, q = i & 15;
      async_copy16(&Ks[t][q * 8], &Kbase[(size_t)(t0 + t) * HD_ + q * 8]);
    }
#pragma unroll
    for (int i = tid; i < 128 * 8; i += 256) {
      const int d = i >> 3, q = i & 7;
      async_copy16(&Vs[d][q * 8], &Vbase[(size_t)d * S_ + t0 + q * 8]);
    }
    wait_async0();
    __syncthreads();

    // S = Q * K^T  (16 x 64), 4 n-frags x 4 k-slices
    v8f sc[4];
#pragma unroll
    for (int n = 0; n < 4; ++n) {
      sc[n] = {};
#pragma unroll
      for (int kk = 0; kk < 4; ++kk)
        sc[n] = __builtin_amdgcn_wmma_f32_16x16x32_bf16(
            false, qf[kk], false, load_frag(&Ks[n * 16][kk * 32], 128, lane),
            (short)0, sc[n], false, false);
    }

    // online softmax (rows v + 8g, columns across 16 lanes of the group)
    float mnew[8];
#pragma unroll
    for (int v = 0; v < 8; ++v) {
      float mv = -3.402823466e38f;
#pragma unroll
      for (int n = 0; n < 4; ++n) mv = fmaxf(mv, sc[n][v] * scale);
#pragma unroll
      for (int off = 1; off < 16; off <<= 1)
        mv = fmaxf(mv, __shfl_xor(mv, off, 32));
      mnew[v] = fmaxf(mrow[v], mv);
    }
#pragma unroll
    for (int v = 0; v < 8; ++v) {
      const float alpha = __expf(mrow[v] - mnew[v]);
      lrow[v] *= alpha;
#pragma unroll
      for (int j = 0; j < 8; ++j) o[j][v] *= alpha;
      float rs = 0.f;
#pragma unroll
      for (int n = 0; n < 4; ++n) {
        const float p = __expf(sc[n][v] * scale - mnew[v]);
        sc[n][v] = p;
        rs += p;
      }
#pragma unroll
      for (int off = 1; off < 16; off <<= 1) rs += __shfl_xor(rs, off, 32);
      lrow[v] += rs;
      mrow[v] = mnew[v];
    }

    // relayout P (C-layout) -> A-layout via per-wave LDS scratch (no barrier)
    const int cc = lane & 15;
#pragma unroll
    for (int n = 0; n < 4; ++n)
#pragma unroll
      for (int v = 0; v < 8; ++v)
        Ps[wave][v + 8 * g][n * 16 + cc] = f32_to_bf16(sc[n][v]);

    // O += P * V   (16 x 128), 2 k-slices x 8 hd-frags
#pragma unroll
    for (int kk = 0; kk < 2; ++kk) {
      const v16bf pa = load_frag(&Ps[wave][0][kk * 32], 64, lane);
#pragma unroll
      for (int j = 0; j < 8; ++j)
        o[j] = __builtin_amdgcn_wmma_f32_16x16x32_bf16(
            false, pa, false, load_frag(&Vs[j * 16][kk * 32], 64, lane),
            (short)0, o[j], false, false);
    }
    __syncthreads();
  }

  // epilogue: normalize and store (B,S,H,HD) bf16 (feeds final GEMM directly)
  const int cc = lane & 15;
#pragma unroll
  for (int j = 0; j < 8; ++j)
#pragma unroll
    for (int v = 0; v < 8; ++v) {
      const int qrow = qt * 128 + wave * 16 + v + 8 * g;
      Oout[(((size_t)b * S_ + qrow) * H_ + h) * HD_ + j * 16 + cc] =
          f32_to_bf16(o[j][v] / lrow[v]);
    }
}

// ---------------------------------------------------------------------------
// Host-side launch sequence
// ---------------------------------------------------------------------------
extern "C" void kernel_launch(void* const* d_in, const int* in_sizes, int n_in,
                              void* d_out, int out_size, void* d_ws, size_t ws_size,
                              hipStream_t stream) {
  const float* x  = (const float*)d_in[0];
  const float* fc = (const float*)d_in[1];
  const float* fs = (const float*)d_in[2];
  const float* wq = (const float*)d_in[3];
  const float* wk = (const float*)d_in[4];
  const float* wv = (const float*)d_in[5];
  const float* wo = (const float*)d_in[6];
  // cache_k / cache_v / start_pos unused (start_pos == 0, full-length softmax)
  float* out = (float*)d_out;

  char* ws = (char*)d_ws;
  unsigned short* xb   = (unsigned short*)(ws);                        // 16.8MB
  unsigned short* wqT  = (unsigned short*)(ws + (size_t)16777216);     //  8.4MB
  unsigned short* wkT  = (unsigned short*)(ws + (size_t)25165824);     //  2.1MB
  unsigned short* wvT  = (unsigned short*)(ws + (size_t)27262976);     //  2.1MB
  unsigned short* woT  = (unsigned short*)(ws + (size_t)29360128);     //  8.4MB
  unsigned short* qpre = (unsigned short*)(ws + (size_t)37748736);     // 16.8MB
  unsigned short* kpre = (unsigned short*)(ws + (size_t)54525952);     //  4.2MB
  unsigned short* vpre = (unsigned short*)(ws + (size_t)58720256);     //  4.2MB
  unsigned short* Qb   = (unsigned short*)(ws + (size_t)62914560);     // 16.8MB
  unsigned short* Kb   = (unsigned short*)(ws + (size_t)79691776);     //  4.2MB
  unsigned short* Vtb  = (unsigned short*)(ws + (size_t)83886080);     //  4.2MB
  unsigned short* attnB = xb;  // alias: x(bf16) dead after the QKV GEMMs

  // 0) one-time bf16 conversions / weight transposes
  convert_f32_bf16_vec4<<<(MTOT * D_ / 4) / 256, 256, 0, stream>>>(x, xb);
  transpose_w_bf16<<<dim3(D_ / 32, D_ / 32), dim3(32, 8), 0, stream>>>(wq, wqT, D_, D_);
  transpose_w_bf16<<<dim3(KVH_ * HD_ / 32, D_ / 32), dim3(32, 8), 0, stream>>>(wk, wkT, D_, KVH_ * HD_);
  transpose_w_bf16<<<dim3(KVH_ * HD_ / 32, D_ / 32), dim3(32, 8), 0, stream>>>(wv, wvT, D_, KVH_ * HD_);
  transpose_w_bf16<<<dim3(D_ / 32, D_ / 32), dim3(32, 8), 0, stream>>>(wo, woT, H_ * HD_, D_);

  // 1) projections (bf16 WMMA GEMMs, bf16 out)
  gemm_bf16wmma<true><<<dim3(H_ * HD_ / 128, MTOT / 256), 256, 0, stream>>>(
      xb, wqT, qpre, MTOT, H_ * HD_, D_);
  gemm_bf16wmma<true><<<dim3(KVH_ * HD_ / 128, MTOT / 256), 256, 0, stream>>>(
      xb, wkT, kpre, MTOT, KVH_ * HD_, D_);
  gemm_bf16wmma<true><<<dim3(KVH_ * HD_ / 128, MTOT / 256), 256, 0, stream>>>(
      xb, wvT, vpre, MTOT, KVH_ * HD_, D_);

  // 2) RoPE + relayout for attention
  rope_q_kernel<<<(B_ * S_ * H_ * (HD_ / 2)) / 256, 256, 0, stream>>>(qpre, fc, fs, Qb);
  rope_k_kernel<<<(B_ * S_ * KVH_ * (HD_ / 2)) / 256, 256, 0, stream>>>(kpre, fc, fs, Kb);
  v_transpose_kernel<<<(B_ * S_ * KVH_ * HD_) / 256, 256, 0, stream>>>(vpre, Vtb);

  // 3) attention (bf16 out, laid out as A-matrix of the final GEMM)
  attn_kernel<<<dim3(S_ / 128, H_, B_), 256, 0, stream>>>(Qb, Kb, Vtb, attnB);

  // 4) output projection (f32 out)
  gemm_bf16wmma<false><<<dim3(D_ / 128, MTOT / 256), 256, 0, stream>>>(
      attnB, woT, out, MTOT, D_, H_ * HD_);
}
